// SwinWindowAttention_6124623364408
// MI455X (gfx1250) — compile-verified, matrix operand order checked by reading, not run on hardware
//
#include <hip/hip_runtime.h>
#include <hip/hip_bf16.h>

// ---------------------------------------------------------------------------
// Swin 3D window attention for gfx1250 (MI455X), bf16 WMMA path.
//   B=64 windows, N=343 tokens, C=512, H=16 heads, hd=32, NW=8 masks.
// Pipeline:
//   k0: biasH[h,i,j] = bias_table[rel_index[i,j], h]            (f32, 7.5 MB)
//   k1: qkv = x @ w_qkv + b_qkv -> q(scaled)/k/v bf16 [B,H,N,hd]
//   k2: per (b,h): S = qK^T + biasH + mask; softmax; O = P @ v -> bf16 [B,N,C]
//   k3: out = O @ w_proj + b_proj (f32)
// All WMMA operands are staged in LDS in fragment-native (K-contiguous) order
// so every v16bf fragment is exactly two aligned ds_load_b128 per lane.
// ---------------------------------------------------------------------------

typedef __bf16 bf16;
typedef __attribute__((ext_vector_type(16))) __bf16 v16bf;
typedef __attribute__((ext_vector_type(8)))  __bf16 v8bf;
typedef __attribute__((ext_vector_type(8)))  float  v8f;

#define SEQ     343
#define SEQP    352          // 11*32
#define CDIM    512
#define HD      32
#define NHEADS  16
#define NN      (SEQ * SEQ)  // 117649
#define QSCALE  0.17677669529663687f   // 32^-0.5

// Per-lane fragment = rows of 16 bf16: k in [8h,8h+8) and [16+8h,16+8h+8).
// With K contiguous in LDS this is two aligned 16B loads.
__device__ __forceinline__ v16bf load_frag(const bf16* p) {
  v8bf lo = *(const v8bf*)(p);
  v8bf hi = *(const v8bf*)(p + 16);
  return __builtin_shufflevector(lo, hi, 0, 1, 2, 3, 4, 5, 6, 7,
                                 8, 9, 10, 11, 12, 13, 14, 15);
}

// ---------------------------------------------------------------------------
// Kernel 0: expand relative-position bias:  biasH[h*NN + i*SEQ + j]
// ---------------------------------------------------------------------------
__global__ __launch_bounds__(256) void bias_expand_kernel(
    const float* __restrict__ bias_table, const int* __restrict__ rel_index,
    float* __restrict__ biasH)
{
  int e = blockIdx.x * 256 + threadIdx.x;
  if (e >= NHEADS * NN) return;
  int h = e / NN;
  int r = e - h * NN;
  biasH[e] = bias_table[rel_index[r] * NHEADS + h];
}

// ---------------------------------------------------------------------------
// Kernel 1: QKV projection. A = x [21952 x 512] f32, B = w_qkv [512 x 1536].
// Block: 256 thr = 8 waves; tile 64(M) x 64(cols); waves 4x2, 2 col-tiles each.
// ---------------------------------------------------------------------------
__global__ __launch_bounds__(256) void qkv_kernel(
    const float* __restrict__ x, const float* __restrict__ wqkv,
    const float* __restrict__ bqkv,
    bf16* __restrict__ qws, bf16* __restrict__ kws, bf16* __restrict__ vws)
{
  __shared__ bf16 As[64 * 40];    // [row][k]   k contiguous, stride 40
  __shared__ bf16 Bst[64 * 40];   // [col][k]   transposed, k contiguous

  const int tid  = threadIdx.x;
  const int lane = tid & 31;
  const int wave = tid >> 5;
  const int half = lane >> 4;
  const int ln   = lane & 15;
  const int cg   = blockIdx.x;   // 0..23
  const int g    = blockIdx.y;   // 0..342
  const int wm   = wave & 3;
  const int wn   = wave >> 2;

  v8f acc0 = {}; v8f acc1 = {};

  for (int k0 = 0; k0 < CDIM; k0 += 32) {
    { // stage A: 64x32 f32 -> bf16, row-major (k contiguous)
      int base = tid * 8, row = base >> 5, col = base & 31;
      const float* src = x + (size_t)(g * 64 + row) * CDIM + k0 + col;
      float4 f0 = *(const float4*)(src);
      float4 f1 = *(const float4*)(src + 4);
      bf16* d = As + row * 40 + col;
      d[0] = (bf16)f0.x; d[1] = (bf16)f0.y; d[2] = (bf16)f0.z; d[3] = (bf16)f0.w;
      d[4] = (bf16)f1.x; d[5] = (bf16)f1.y; d[6] = (bf16)f1.z; d[7] = (bf16)f1.w;
    }
    { // stage B transposed: read 8 consecutive cols of row k, scatter to [col][k]
      int k = tid >> 3, n0 = (tid & 7) * 8;
      const float* src = wqkv + (size_t)(k0 + k) * 1536 + cg * 64 + n0;
      float4 f0 = *(const float4*)(src);
      float4 f1 = *(const float4*)(src + 4);
      Bst[(n0 + 0) * 40 + k] = (bf16)f0.x; Bst[(n0 + 1) * 40 + k] = (bf16)f0.y;
      Bst[(n0 + 2) * 40 + k] = (bf16)f0.z; Bst[(n0 + 3) * 40 + k] = (bf16)f0.w;
      Bst[(n0 + 4) * 40 + k] = (bf16)f1.x; Bst[(n0 + 5) * 40 + k] = (bf16)f1.y;
      Bst[(n0 + 6) * 40 + k] = (bf16)f1.z; Bst[(n0 + 7) * 40 + k] = (bf16)f1.w;
    }
    __syncthreads();

    v16bf a  = load_frag(As  + (wm * 16 + ln) * 40 + (half << 3));
    v16bf b0 = load_frag(Bst + (wn * 32 + ln) * 40 + (half << 3));
    v16bf b1 = load_frag(Bst + (wn * 32 + 16 + ln) * 40 + (half << 3));
    acc0 = __builtin_amdgcn_wmma_f32_16x16x32_bf16(false, a, false, b0, (short)0, acc0, false, false);
    acc1 = __builtin_amdgcn_wmma_f32_16x16x32_bf16(false, a, false, b1, (short)0, acc1, false, false);
    __syncthreads();
  }

  // Epilogue: row decomposition hoisted out of the column loop.
  int brow[8], nrow[8];
#pragma unroll
  for (int r = 0; r < 8; ++r) {
    int m = g * 64 + wm * 16 + r + (half << 3);
    brow[r] = m / SEQ;
    nrow[r] = m - brow[r] * SEQ;
  }
#pragma unroll
  for (int ct = 0; ct < 2; ++ct) {
    v8f acc = ct ? acc1 : acc0;
    int c   = cg * 64 + wn * 32 + ct * 16 + ln;
    int sel = c >> 9;
    int cc  = c & 511;
    int h   = cc >> 5, d = cc & 31;
    float sc   = (sel == 0) ? QSCALE : 1.0f;
    float badd = bqkv[c];
    bf16* dst = (sel == 0) ? qws : ((sel == 1) ? kws : vws);
#pragma unroll
    for (int r = 0; r < 8; ++r) {
      float v = (acc[r] + badd) * sc;
      dst[(size_t)((brow[r] * NHEADS + h) * SEQ + nrow[r]) * HD + d] = (bf16)v;
    }
  }
}

// ---------------------------------------------------------------------------
// Kernel 2: attention for one (b,h).
//   LDS: qs [352][40] + ks [352][40] bf16, vst [32][360] bf16 (transposed),
//        S [176][344] f32  -> 321,536 B total (<= 320 KB WGP LDS).
//   Softmax narrows P to bf16 in place (stride 688 bf16 per row, zero-padded
//   to k=352) so the P@V A-fragments are unguarded vector loads.
// ---------------------------------------------------------------------------
__global__ __launch_bounds__(256) void attn_kernel(
    const bf16* __restrict__ qws, const bf16* __restrict__ kws,
    const bf16* __restrict__ vws,
    const float* __restrict__ biasH, const float* __restrict__ mask,
    bf16* __restrict__ ao)
{
  extern __shared__ __align__(16) char smem[];
  bf16*  qs  = (bf16*)smem;                 // [352][40]
  bf16*  ks  = qs + SEQP * 40;              // [352][40]
  bf16*  vst = ks + SEQP * 40;              // [32][360]  (d-major, k contiguous)
  float* S   = (float*)(vst + 32 * 360);    // [176][344]

  const int tid    = threadIdx.x;
  const int lane   = tid & 31;
  const int wave   = tid >> 5;
  const int half_l = lane >> 4;
  const int ln     = lane & 15;
  const int bh = blockIdx.x;
  const int b  = bh >> 4;
  const int h  = bh & 15;

  const size_t gbase = (size_t)(b * NHEADS + h) * SEQ * HD;
  const uint4* gq = (const uint4*)(qws + gbase);
  const uint4* gk = (const uint4*)(kws + gbase);
  const uint4* gv = (const uint4*)(vws + gbase);
  const uint4 z = {0u, 0u, 0u, 0u};
  // q, k row-major (padded rows zeroed); v transposed into vst.
  for (int i = tid; i < SEQP * 4; i += 256) {
    int row = i >> 2, seg = i & 3;
    uint4 vq = z, vk = z, vv = z;
    if (row < SEQ) { int s = row * 4 + seg; vq = gq[s]; vk = gk[s]; vv = gv[s]; }
    *((uint4*)(qs + row * 40) + seg) = vq;
    *((uint4*)(ks + row * 40) + seg) = vk;
    const bf16* pv = (const bf16*)&vv;
    int d0 = seg * 8;
#pragma unroll
    for (int j = 0; j < 8; ++j) vst[(d0 + j) * 360 + row] = pv[j];
  }
  __syncthreads();

  const float* maskw = mask + (size_t)(b & 7) * NN;
  const float* biasw = biasH + (size_t)h * NN;

  for (int half = 0; half < 2; ++half) {
    const int row0 = half * 176;
    const int rows = half ? (SEQ - 176) : 176;

    // ---- S = q @ k^T (+bias +mask); A fragment reused across nt tiles ----
    for (int mt = 0; mt < 11; ++mt) {
      v16bf a = load_frag(qs + (row0 + mt * 16 + ln) * 40 + (half_l << 3));
      for (int nt = wave; nt < 22; nt += 8) {
        v16bf bb = load_frag(ks + (nt * 16 + ln) * 40 + (half_l << 3));
        v8f acc = {};
        acc = __builtin_amdgcn_wmma_f32_16x16x32_bf16(false, a, false, bb, (short)0, acc, false, false);
        int n = nt * 16 + ln;
#pragma unroll
        for (int r = 0; r < 8; ++r) {
          int ml = mt * 16 + r + (half_l << 3);
          int mg = row0 + ml;
          if (mg < SEQ && n < SEQ) {
            int off = mg * SEQ + n;
            S[ml * 344 + n] = acc[r] + biasw[off] + maskw[off];
          }
        }
      }
    }
    __syncthreads();

    // ---- softmax (f32) + in-place narrowing to bf16 P, zero-padded ----
    if (tid < rows) {
      float* row = S + tid * 344;
      float mx = -3.4e38f;
      for (int j = 0; j < SEQ; ++j) mx = fmaxf(mx, row[j]);
      float sum = 0.f;
      for (int j = 0; j < SEQ; ++j) { float e = __expf(row[j] - mx); row[j] = e; sum += e; }
      float inv = 1.f / sum;
      bf16* pb = (bf16*)row;           // ascending narrowing: read f32[j] before
      for (int j = 0; j < SEQ; ++j) {  // its low half is overwritten (2j < 4j)
        float p = row[j] * inv;
        pb[j] = (bf16)p;
      }
      for (int j = SEQ; j < SEQP; ++j) pb[j] = (bf16)0.0f;
    }
    __syncthreads();

    // ---- O = P @ v : 11 WMMA along K=352, all-vector fragment loads ----
    for (int t = wave; t < 22; t += 8) {
      int mt = t >> 1, nt = t & 1;
      int dcol = nt * 16 + ln;
      const bf16* prow = (const bf16*)S + (size_t)(mt * 16 + ln) * 688 + (half_l << 3);
      const bf16* vrow = vst + dcol * 360 + (half_l << 3);
      v8f acc = {};
      for (int kt = 0; kt < 11; ++kt) {
        v16bf a  = load_frag(prow + kt * 32);
        v16bf bb = load_frag(vrow + kt * 32);
        acc = __builtin_amdgcn_wmma_f32_16x16x32_bf16(false, a, false, bb, (short)0, acc, false, false);
      }
#pragma unroll
      for (int r = 0; r < 8; ++r) {
        int mg = row0 + mt * 16 + r + (half_l << 3);
        if (mg < SEQ)
          ao[(size_t)(b * SEQ + mg) * CDIM + h * HD + dcol] = (bf16)acc[r];
      }
    }
    __syncthreads();
  }
}

// ---------------------------------------------------------------------------
// Kernel 3: output projection. A = ao [21952 x 512] bf16, B = w_proj [512x512].
// ---------------------------------------------------------------------------
__global__ __launch_bounds__(256) void proj_kernel(
    const bf16* __restrict__ ao, const float* __restrict__ wproj,
    const float* __restrict__ bproj, float* __restrict__ out)
{
  __shared__ bf16 As[64 * 40];
  __shared__ bf16 Bst[64 * 40];

  const int tid  = threadIdx.x;
  const int lane = tid & 31;
  const int wave = tid >> 5;
  const int half = lane >> 4;
  const int ln   = lane & 15;
  const int cg   = blockIdx.x;   // 0..7
  const int g    = blockIdx.y;   // 0..342
  const int wm   = wave & 3;
  const int wn   = wave >> 2;

  v8f acc0 = {}; v8f acc1 = {};

  for (int k0 = 0; k0 < CDIM; k0 += 32) {
    { // stage A: 64x32 bf16 vector copy
      int base = tid * 8, row = base >> 5, col = base & 31;
      const uint4 v = *(const uint4*)(ao + (size_t)(g * 64 + row) * CDIM + k0 + col);
      *(uint4*)(As + row * 40 + col) = v;
    }
    { // stage B transposed
      int k = tid >> 3, n0 = (tid & 7) * 8;
      const float* src = wproj + (size_t)(k0 + k) * CDIM + cg * 64 + n0;
      float4 f0 = *(const float4*)(src);
      float4 f1 = *(const float4*)(src + 4);
      Bst[(n0 + 0) * 40 + k] = (bf16)f0.x; Bst[(n0 + 1) * 40 + k] = (bf16)f0.y;
      Bst[(n0 + 2) * 40 + k] = (bf16)f0.z; Bst[(n0 + 3) * 40 + k] = (bf16)f0.w;
      Bst[(n0 + 4) * 40 + k] = (bf16)f1.x; Bst[(n0 + 5) * 40 + k] = (bf16)f1.y;
      Bst[(n0 + 6) * 40 + k] = (bf16)f1.z; Bst[(n0 + 7) * 40 + k] = (bf16)f1.w;
    }
    __syncthreads();

    v16bf a  = load_frag(As  + (wm * 16 + ln) * 40 + (half << 3));
    v16bf b0 = load_frag(Bst + (wn * 32 + ln) * 40 + (half << 3));
    v16bf b1 = load_frag(Bst + (wn * 32 + 16 + ln) * 40 + (half << 3));
    acc0 = __builtin_amdgcn_wmma_f32_16x16x32_bf16(false, a, false, b0, (short)0, acc0, false, false);
    acc1 = __builtin_amdgcn_wmma_f32_16x16x32_bf16(false, a, false, b1, (short)0, acc1, false, false);
    __syncthreads();
  }

#pragma unroll
  for (int ct = 0; ct < 2; ++ct) {
    v8f acc = ct ? acc1 : acc0;
    int c = cg * 64 + wn * 32 + ct * 16 + ln;
    float badd = bproj[c];
#pragma unroll
    for (int r = 0; r < 8; ++r) {
      int m = g * 64 + wm * 16 + r + (half << 3);
      out[(size_t)m * CDIM + c] = acc[r] + badd;
    }
  }
}

// ---------------------------------------------------------------------------
extern "C" void kernel_launch(void* const* d_in, const int* in_sizes, int n_in,
                              void* d_out, int out_size, void* d_ws, size_t ws_size,
                              hipStream_t stream) {
  const float* x          = (const float*)d_in[0];
  const float* mask       = (const float*)d_in[1];
  const float* w_qkv      = (const float*)d_in[2];
  const float* b_qkv      = (const float*)d_in[3];
  const float* w_proj     = (const float*)d_in[4];
  const float* b_proj     = (const float*)d_in[5];
  const float* bias_table = (const float*)d_in[6];
  const int*   rel_index  = (const int*)d_in[7];

  const size_t QKV_ELEMS = (size_t)64 * NHEADS * SEQ * HD;  // 11,239,424
  bf16*  qws   = (bf16*)d_ws;
  bf16*  kws   = qws + QKV_ELEMS;
  bf16*  vws   = kws + QKV_ELEMS;
  bf16*  aows  = vws + QKV_ELEMS;
  float* biasH = (float*)(aows + QKV_ELEMS);                // 16*NN f32

  bias_expand_kernel<<<(NHEADS * NN + 255) / 256, 256, 0, stream>>>(
      bias_table, rel_index, biasH);

  qkv_kernel<<<dim3(24, 343), 256, 0, stream>>>(x, w_qkv, b_qkv, qws, kws, vws);

  const size_t smem = (size_t)2 * SEQP * 40 * sizeof(bf16)   // qs, ks
                    + (size_t)32 * 360 * sizeof(bf16)        // vst
                    + (size_t)176 * 344 * sizeof(float);     // S  -> 321,536 B
  attn_kernel<<<dim3(1024), 256, smem, stream>>>(qws, kws, vws,
                                                 biasH, mask, aows);

  proj_kernel<<<dim3(8, 343), 256, 0, stream>>>(aows, w_proj, b_proj, (float*)d_out);
}
